// TPAAttention_6863357739640
// MI455X (gfx1250) — matrix operand order repaired and checked by reading, not compiled, not run on hardware
//
#include <hip/hip_runtime.h>

// ---------------------------------------------------------------------------
// TPA attention for MI455X (gfx1250, wave32, WMMA + Tensor Data Mover).
// Pipeline: bf16 convert -> WMMA projection GEMMs (TDM-staged B tiles)
//           -> rank-combine + RoPE -> WMMA flash attention (TDM-staged K/V,
//              softcap, causal, online softmax) -> WMMA output GEMM.
// ---------------------------------------------------------------------------

#define S_LEN  2048
#define HID    2048
#define NH     16
#define NKVH   8
#define HD     128
#define QR     6
#define KR     2
#define VR     2
#define SOFTCAP 50.0f
#define NEGINF (-1000000000.0f)

#if __has_builtin(__builtin_amdgcn_tensor_load_to_lds) && __has_builtin(__builtin_amdgcn_s_wait_tensorcnt)
#define HAVE_TDM 1
#else
#define HAVE_TDM 0
#endif

typedef __attribute__((ext_vector_type(16))) __bf16 v16bf;
typedef __attribute__((ext_vector_type(8)))  __bf16 v8bf;
typedef __attribute__((ext_vector_type(8)))  float  v8f;
typedef __attribute__((ext_vector_type(4)))  unsigned int u32x4;
typedef __attribute__((ext_vector_type(8)))  int i32x8;
typedef __attribute__((ext_vector_type(4)))  int i32x4;

__device__ __forceinline__ __bf16 f2bf(float x) {
  unsigned u = __builtin_bit_cast(unsigned, x);
  u += 0x7FFFu + ((u >> 16) & 1u);           // round-to-nearest-even
  unsigned short h = (unsigned short)(u >> 16);
  return __builtin_bit_cast(__bf16, h);
}

// A-operand fragment (16x32 bf16), row-major src, ld in elements.
// Lanes 0-15: row=lane, K = 0..7 & 16..23 ; lanes 16-31: K = 8..15 & 24..31.
__device__ __forceinline__ v16bf load_a_frag(const __bf16* base, int ld) {
  int lane = threadIdx.x & 31;
  int r  = lane & 15;
  int kb = (lane >> 4) << 3;                 // 0 or 8
  const __bf16* p = base + (size_t)r * ld + kb;
  v8bf lo = *(const v8bf*)p;
  v8bf hi = *(const v8bf*)(p + 16);
  return __builtin_shufflevector(lo, hi, 0,1,2,3,4,5,6,7,8,9,10,11,12,13,14,15);
}

// B-operand fragment (32x16 bf16) from an N x K row-major (transposed) matrix.
// Lanes 0-15: col n=lane, K = 0..7 & 16..23 ; lanes 16-31: K = 8..15 & 24..31.
__device__ __forceinline__ v16bf load_b_frag(const __bf16* baseT, int ld) {
  int lane = threadIdx.x & 31;
  int n  = lane & 15;
  int kb = (lane >> 4) << 3;                 // 0 or 8
  const __bf16* p = baseT + (size_t)n * ld + kb;
  v8bf lo = *(const v8bf*)p;
  v8bf hi = *(const v8bf*)(p + 16);
  return __builtin_shufflevector(lo, hi, 0,1,2,3,4,5,6,7,8,9,10,11,12,13,14,15);
}

#if HAVE_TDM
// 2D TDM descriptor per CDNA5 ISA 8.3/8.4: tile (tile_d0 contiguous x tile_d1
// rows, row stride stride0 elements, 2-byte data) -> LDS at lds_off.
// 6-arg builtin (clang-23 flavor): groups 2/3/4 zero => tensor <= 2D.
__device__ __forceinline__ void tdm_load_2d(const void* gaddr, unsigned lds_off,
                                            unsigned tile_d0, unsigned tile_d1,
                                            unsigned tensor_d0, unsigned tensor_d1,
                                            unsigned long stride0) {
  unsigned long ga = (unsigned long)gaddr;
  u32x4 g0;
  g0[0] = 1u;                                          // count=1, user descriptor
  g0[1] = lds_off;                                     // lds_addr (bytes)
  g0[2] = (unsigned)(ga & 0xFFFFFFFFu);                // global_addr[31:0]
  g0[3] = (unsigned)((ga >> 32) & 0x01FFFFFFu) | (2u << 30);  // addr[56:32], type=2
  i32x8 g1;
  g1[0] = (int)(1u << 16);                             // wg_mask=0, data_size=1 (2B)
  g1[1] = (int)((tensor_d0 & 0xFFFFu) << 16);          // tensor_dim0[15:0]
  g1[2] = (int)((tensor_d0 >> 16) | ((tensor_d1 & 0xFFFFu) << 16));
  g1[3] = (int)((tensor_d1 >> 16) | (tile_d0 << 16));  // tile_dim0
  g1[4] = (int)tile_d1;                                // tile_dim1, tile_dim2=0
  g1[5] = (int)(stride0 & 0xFFFFFFFFu);                // tensor_dim0_stride[31:0]
  g1[6] = (int)((stride0 >> 32) & 0xFFFFu);            // stride0[47:32], stride1=0
  g1[7] = 0;
  i32x4 z4 = {0, 0, 0, 0};
  i32x8 z8 = {0, 0, 0, 0, 0, 0, 0, 0};
  __builtin_amdgcn_tensor_load_to_lds(g0, g1, z4, z4, z8, 0);
}
#endif

// ------------------------------- converters --------------------------------
__global__ void k_f32_to_bf16(const float* __restrict__ in, __bf16* __restrict__ out, int n) {
  int i = blockIdx.x * blockDim.x + threadIdx.x;
  if (i < n) out[i] = f2bf(in[i]);
}

// in: K x N (row-major f32) -> out: N x K (row-major bf16)
__global__ void k_conv_transpose(const float* __restrict__ in, __bf16* __restrict__ out,
                                 int Kd, int Nd) {
  int i = blockIdx.x * blockDim.x + threadIdx.x;
  if (i >= Kd * Nd) return;
  int k = i / Nd, n = i % Nd;
  out[(size_t)n * Kd + k] = f2bf(in[i]);
}

// ----------------------------- WMMA GEMM -----------------------------------
// C[M,N] = A[M,K](bf16) * Bt[N,K](bf16)^T, f32 out. 8 waves/block (rows),
// wave = 16 x (16*NT) strip. B tile double-buffered in LDS via TDM DMA.
template <int NT>
__global__ void __launch_bounds__(256) k_gemm_bf16(const __bf16* __restrict__ A,
                                                   const __bf16* __restrict__ Bt,
                                                   float* __restrict__ C,
                                                   int n_base, int M, int N, int K) {
  int wave = threadIdx.x >> 5, lane = threadIdx.x & 31;
  int m0 = blockIdx.y * 128 + wave * 16;
  int n0 = n_base + blockIdx.x * (16 * NT);
  v8f acc[NT] = {};

#if HAVE_TDM
  __shared__ __bf16 lds_b[2][16 * NT][32];
  const int niter = K / 32;
  const __bf16* bt0 = Bt + (size_t)n0 * K;
  if (threadIdx.x < 32)                                 // wave 0 drives the DMA
    tdm_load_2d(bt0, (unsigned)(uintptr_t)&lds_b[0][0][0],
                32, 16 * NT, (unsigned)K, 16 * NT, (unsigned long)K);
  for (int i = 0; i < niter; ++i) {
    int k0 = i * 32;
    __syncthreads();                                    // prior reads of next buf done
    if (threadIdx.x < 32) {
      if (i + 1 < niter) {
        tdm_load_2d(bt0 + (k0 + 32), (unsigned)(uintptr_t)&lds_b[(i + 1) & 1][0][0],
                    32, 16 * NT, (unsigned)K, 16 * NT, (unsigned long)K);
        __builtin_amdgcn_s_wait_tensorcnt(1);           // current tile landed
      } else {
        __builtin_amdgcn_s_wait_tensorcnt(0);
      }
    }
    __syncthreads();                                    // tile visible to all waves
    const __bf16* ap = A + (size_t)m0 * K + k0;
    if (k0 + 128 < K) __builtin_prefetch(ap + 128, 0, 1);  // global_prefetch_b8
    v16bf a = load_a_frag(ap, K);
#pragma unroll
    for (int j = 0; j < NT; ++j) {
      v16bf b = load_b_frag(&lds_b[i & 1][16 * j][0], 32);
      acc[j] = __builtin_amdgcn_wmma_f32_16x16x32_bf16(false, a, false, b,
                                                       (short)0, acc[j], false, false);
    }
  }
#else
  for (int k0 = 0; k0 < K; k0 += 32) {
    const __bf16* ap = A + (size_t)m0 * K + k0;
    v16bf a = load_a_frag(ap, K);
#pragma unroll
    for (int j = 0; j < NT; ++j) {
      v16bf b = load_b_frag(Bt + (size_t)(n0 + 16 * j) * K + k0, K);
      acc[j] = __builtin_amdgcn_wmma_f32_16x16x32_bf16(false, a, false, b,
                                                       (short)0, acc[j], false, false);
    }
  }
#endif
  int r_hi = (lane >> 4) << 3, col = lane & 15;
#pragma unroll
  for (int j = 0; j < NT; ++j) {
#pragma unroll
    for (int r = 0; r < 8; ++r)
      C[(size_t)(m0 + r + r_hi) * N + n0 + 16 * j + col] = acc[j][r];
  }
}

// ------------------------ rank combine + RoPE ------------------------------
__global__ void k_combine_q(const float* __restrict__ Aq, const float* __restrict__ Bq,
                            const float* __restrict__ cs, const float* __restrict__ sn,
                            __bf16* __restrict__ Q) {
  int idx = blockIdx.x * blockDim.x + threadIdx.x;       // (q, h, i<64)
  if (idx >= S_LEN * NH * (HD / 2)) return;
  int i = idx & 63, h = (idx >> 6) & (NH - 1), q = idx >> 10;
  float xe = 0.f, xo = 0.f;
  const float* a = Aq + (size_t)q * (NH * QR) + h * QR;
  const float* b = Bq + (size_t)q * (QR * HD);
#pragma unroll
  for (int r = 0; r < QR; ++r) {
    float ar = a[r];
    xe += ar * b[r * HD + 2 * i];
    xo += ar * b[r * HD + 2 * i + 1];
  }
  float c = cs[q * 64 + i], s = sn[q * 64 + i];
  const float scaling = 0.08838834764831845f;            // 128^-0.5
  __bf16* o = Q + ((size_t)h * S_LEN + q) * HD + 2 * i;
  o[0] = f2bf((xe * c - xo * s) * scaling);
  o[1] = f2bf((xe * s + xo * c) * scaling);
}

__global__ void k_combine_k(const float* __restrict__ Ak, const float* __restrict__ Bk,
                            const float* __restrict__ cs, const float* __restrict__ sn,
                            __bf16* __restrict__ Kt) {
  int idx = blockIdx.x * blockDim.x + threadIdx.x;       // (k, h, i<64)
  if (idx >= S_LEN * NKVH * (HD / 2)) return;
  int i = idx & 63, h = (idx >> 6) & (NKVH - 1), k = idx >> 9;
  const float* a = Ak + (size_t)k * (NKVH * KR) + h * KR;
  const float* b = Bk + (size_t)k * (KR * HD);
  float xe = a[0] * b[2 * i]     + a[1] * b[HD + 2 * i];
  float xo = a[0] * b[2 * i + 1] + a[1] * b[HD + 2 * i + 1];
  float c = cs[k * 64 + i], s = sn[k * 64 + i];
  __bf16* o = Kt + ((size_t)h * S_LEN + k) * HD + 2 * i;
  o[0] = f2bf(xe * c - xo * s);
  o[1] = f2bf(xe * s + xo * c);
}

__global__ void k_combine_v(const float* __restrict__ Av, const float* __restrict__ Bv,
                            __bf16* __restrict__ Vt) {
  int idx = blockIdx.x * blockDim.x + threadIdx.x;       // (k, h, d)
  if (idx >= S_LEN * NKVH * HD) return;
  int d = idx & 127, h = (idx >> 7) & (NKVH - 1), k = idx >> 10;
  const float* a = Av + (size_t)k * (NKVH * VR) + h * VR;
  const float* b = Bv + (size_t)k * (VR * HD);
  Vt[((size_t)h * HD + d) * S_LEN + k] = f2bf(a[0] * b[d] + a[1] * b[HD + d]);
}

// -------------------------- flash attention --------------------------------
// Block: 128 q-rows x 1 head (8 waves, wave = 16 rows). 32-key inner blocks.
// K/V slabs shared by all 8 waves -> double-buffered LDS via TDM DMA.
__global__ void __launch_bounds__(256) k_attention(const __bf16* __restrict__ Q,
                                                   const __bf16* __restrict__ Kt,
                                                   const __bf16* __restrict__ Vt,
                                                   float* __restrict__ O) {
  __shared__ __bf16 pbuf[8][16][32];                     // per-wave P tile (C->A relayout)
#if HAVE_TDM
  __shared__ __bf16 ldsK[2][32][128];                    // [key][d]
  __shared__ __bf16 ldsV[2][128][32];                    // [d][key]
#endif
  int wave = threadIdx.x >> 5, lane = threadIdx.x & 31;
  int h = blockIdx.y, kvh = h >> 1;
  int qb = blockIdx.x * 128;
  int q0 = qb + wave * 16;
  int r_hi = (lane >> 4) << 3, col = lane & 15;

  v16bf qf[4];
  const __bf16* qbase = Q + ((size_t)h * S_LEN + q0) * HD;
#pragma unroll
  for (int c = 0; c < 4; ++c) qf[c] = load_a_frag(qbase + 32 * c, HD);

  v8f o[8] = {};
  float mrow[8], lrow[8];
#pragma unroll
  for (int r = 0; r < 8; ++r) { mrow[r] = -3.0e38f; lrow[r] = 0.f; }

  const __bf16* kbaseP = Kt + (size_t)kvh * S_LEN * HD;  // [key][d], ld=HD
  const __bf16* vbaseP = Vt + (size_t)kvh * HD * S_LEN;  // [d][key], ld=S_LEN
  const int niter = (qb + 128) / 32;                     // causal horizon / 32

#if HAVE_TDM
  if (threadIdx.x < 32) {
    tdm_load_2d(kbaseP, (unsigned)(uintptr_t)&ldsK[0][0][0], 128, 32, 128u, 32u, 128ul);
    tdm_load_2d(vbaseP, (unsigned)(uintptr_t)&ldsV[0][0][0], 32, 128, (unsigned)S_LEN, 128u,
                (unsigned long)S_LEN);
  }
#endif
  for (int it = 0; it < niter; ++it) {
    int kbase = it * 32;
#if HAVE_TDM
    __syncthreads();                                     // prev reads of next buf done
    if (threadIdx.x < 32) {
      if (it + 1 < niter) {
        int nb = (it + 1) & 1;
        tdm_load_2d(kbaseP + (size_t)(kbase + 32) * HD,
                    (unsigned)(uintptr_t)&ldsK[nb][0][0], 128, 32, 128u, 32u, 128ul);
        tdm_load_2d(vbaseP + (kbase + 32),
                    (unsigned)(uintptr_t)&ldsV[nb][0][0], 32, 128, (unsigned)S_LEN, 128u,
                    (unsigned long)S_LEN);
        __builtin_amdgcn_s_wait_tensorcnt(2);            // current K/V landed
      } else {
        __builtin_amdgcn_s_wait_tensorcnt(0);
      }
    }
    __syncthreads();                                     // K/V visible to all waves
    const __bf16* kb0 = &ldsK[it & 1][0][0];
    const int kld = 128;
#else
    const __bf16* kb0 = kbaseP + (size_t)kbase * HD;
    const int kld = HD;
#endif
    // S = Q @ K^T for a 16x32 key slab (SCALING pre-folded in Q)
    v8f s0 = {}, s1 = {};
#pragma unroll
    for (int c = 0; c < 4; ++c) {
      v16bf b0 = load_b_frag(kb0 + 32 * c, kld);
      v16bf b1 = load_b_frag(kb0 + (size_t)16 * kld + 32 * c, kld);
      s0 = __builtin_amdgcn_wmma_f32_16x16x32_bf16(false, qf[c], false, b0, (short)0, s0, false, false);
      s1 = __builtin_amdgcn_wmma_f32_16x16x32_bf16(false, qf[c], false, b1, (short)0, s1, false, false);
    }
    // softcap + causal mask + online softmax (rows live in 16-lane halves)
    float p0[8], p1[8], newm[8];
#pragma unroll
    for (int r = 0; r < 8; ++r) {
      int q_idx = q0 + r + r_hi;
      float x0 = SOFTCAP * tanhf(s0[r] * (1.0f / SOFTCAP));
      float x1 = SOFTCAP * tanhf(s1[r] * (1.0f / SOFTCAP));
      if (kbase + col > q_idx)      x0 += NEGINF;
      if (kbase + 16 + col > q_idx) x1 += NEGINF;
      p0[r] = x0; p1[r] = x1;
      float mx = fmaxf(x0, x1);
#pragma unroll
      for (int off = 8; off >= 1; off >>= 1)
        mx = fmaxf(mx, __shfl_xor(mx, off, 32));
      newm[r] = fmaxf(mrow[r], mx);
    }
#pragma unroll
    for (int r = 0; r < 8; ++r) {
      float e0 = __expf(p0[r] - newm[r]);
      float e1 = __expf(p1[r] - newm[r]);
      p0[r] = e0; p1[r] = e1;
      float srow = e0 + e1;
#pragma unroll
      for (int off = 8; off >= 1; off >>= 1)
        srow += __shfl_xor(srow, off, 32);
      float scale = __expf(mrow[r] - newm[r]);
      lrow[r] = lrow[r] * scale + srow;
      mrow[r] = newm[r];
#pragma unroll
      for (int j = 0; j < 8; ++j) o[j][r] *= scale;
    }
    // C-layout P -> LDS -> A-layout fragment (same-wave DS ops are in-order)
#pragma unroll
    for (int r = 0; r < 8; ++r) {
      pbuf[wave][r + r_hi][col]      = f2bf(p0[r]);
      pbuf[wave][r + r_hi][16 + col] = f2bf(p1[r]);
    }
    v16bf pf = load_a_frag(&pbuf[wave][0][0], 32);
    // O += P @ V
#pragma unroll
    for (int j = 0; j < 8; ++j) {
#if HAVE_TDM
      v16bf bv = load_b_frag(&ldsV[it & 1][16 * j][0], 32);
#else
      v16bf bv = load_b_frag(vbaseP + (size_t)(16 * j) * S_LEN + kbase, S_LEN);
#endif
      o[j] = __builtin_amdgcn_wmma_f32_16x16x32_bf16(false, pf, false, bv, (short)0, o[j], false, false);
    }
  }
  // normalize + write O[q, h*128 + d]
#pragma unroll
  for (int r = 0; r < 8; ++r) {
    float inv = 1.0f / lrow[r];
    int q_idx = q0 + r + r_hi;
#pragma unroll
    for (int j = 0; j < 8; ++j)
      O[(size_t)q_idx * (NH * HD) + h * HD + 16 * j + col] = o[j][r] * inv;
  }
}

// ------------------------------- launch ------------------------------------
static void launch_gemm(const __bf16* A, const __bf16* Bt, float* C,
                        int M, int N, int K, hipStream_t s) {
  int full = N / 64;
  if (full > 0)
    k_gemm_bf16<4><<<dim3(full, M / 128), 256, 0, s>>>(A, Bt, C, 0, M, N, K);
  int rem = (N % 64) / 16, nb = full * 64;
  if (rem == 3) k_gemm_bf16<3><<<dim3(1, M / 128), 256, 0, s>>>(A, Bt, C, nb, M, N, K);
  else if (rem == 2) k_gemm_bf16<2><<<dim3(1, M / 128), 256, 0, s>>>(A, Bt, C, nb, M, N, K);
  else if (rem == 1) k_gemm_bf16<1><<<dim3(1, M / 128), 256, 0, s>>>(A, Bt, C, nb, M, N, K);
}

extern "C" void kernel_launch(void* const* d_in, const int* in_sizes, int n_in,
                              void* d_out, int out_size, void* d_ws, size_t ws_size,
                              hipStream_t stream) {
  const float* hs  = (const float*)d_in[0];
  const float* cs  = (const float*)d_in[1];
  const float* sn  = (const float*)d_in[2];
  // d_in[3] mask, d_in[4] kv_write_indices: recomputed / identity -> unused
  const float* WAq = (const float*)d_in[5];
  const float* WAk = (const float*)d_in[6];
  const float* WAv = (const float*)d_in[7];
  const float* WBq = (const float*)d_in[8];
  const float* WBk = (const float*)d_in[9];
  const float* WBv = (const float*)d_in[10];
  const float* Wo  = (const float*)d_in[11];

  char* ws = (char*)d_ws;
  auto alloc = [&](size_t bytes) -> void* {
    void* p = (void*)ws;
    ws += (bytes + 255) & ~(size_t)255;
    return p;
  };

  __bf16* hsb  = (__bf16*)alloc((size_t)S_LEN * HID * 2);
  __bf16* WAqT = (__bf16*)alloc((size_t)(NH * QR) * HID * 2);
  __bf16* WAkT = (__bf16*)alloc((size_t)(NKVH * KR) * HID * 2);
  __bf16* WAvT = (__bf16*)alloc((size_t)(NKVH * VR) * HID * 2);
  __bf16* WBqT = (__bf16*)alloc((size_t)(QR * HD) * HID * 2);
  __bf16* WBkT = (__bf16*)alloc((size_t)(KR * HD) * HID * 2);
  __bf16* WBvT = (__bf16*)alloc((size_t)(VR * HD) * HID * 2);
  __bf16* WoT  = (__bf16*)alloc((size_t)HID * (NH * HD) * 2);
  float*  Aq   = (float*)alloc((size_t)S_LEN * (NH * QR) * 4);
  float*  Ak   = (float*)alloc((size_t)S_LEN * (NKVH * KR) * 4);
  float*  Av   = (float*)alloc((size_t)S_LEN * (NKVH * VR) * 4);
  float*  Bq   = (float*)alloc((size_t)S_LEN * (QR * HD) * 4);
  float*  Bk   = (float*)alloc((size_t)S_LEN * (KR * HD) * 4);
  float*  Bv   = (float*)alloc((size_t)S_LEN * (VR * HD) * 4);
  __bf16* Qb   = (__bf16*)alloc((size_t)NH * S_LEN * HD * 2);
  __bf16* Kb   = (__bf16*)alloc((size_t)NKVH * S_LEN * HD * 2);
  __bf16* Vtb  = (__bf16*)alloc((size_t)NKVH * HD * S_LEN * 2);
  float*  Obuf = (float*)alloc((size_t)S_LEN * (NH * HD) * 4);
  __bf16* Obf  = (__bf16*)alloc((size_t)S_LEN * (NH * HD) * 2);

  const int T = 256;
  k_f32_to_bf16<<<(S_LEN * HID + T - 1) / T, T, 0, stream>>>(hs, hsb, S_LEN * HID);
  k_conv_transpose<<<(HID * 96  + T - 1) / T, T, 0, stream>>>(WAq, WAqT, HID, 96);
  k_conv_transpose<<<(HID * 16  + T - 1) / T, T, 0, stream>>>(WAk, WAkT, HID, 16);
  k_conv_transpose<<<(HID * 16  + T - 1) / T, T, 0, stream>>>(WAv, WAvT, HID, 16);
  k_conv_transpose<<<(HID * 768 + T - 1) / T, T, 0, stream>>>(WBq, WBqT, HID, 768);
  k_conv_transpose<<<(HID * 256 + T - 1) / T, T, 0, stream>>>(WBk, WBkT, HID, 256);
  k_conv_transpose<<<(HID * 256 + T - 1) / T, T, 0, stream>>>(WBv, WBvT, HID, 256);
  k_conv_transpose<<<(HID * HID + T - 1) / T, T, 0, stream>>>(Wo, WoT, HID, HID);

  launch_gemm(hsb, WAqT, Aq, S_LEN, 96,  HID, stream);
  launch_gemm(hsb, WAkT, Ak, S_LEN, 16,  HID, stream);
  launch_gemm(hsb, WAvT, Av, S_LEN, 16,  HID, stream);
  launch_gemm(hsb, WBqT, Bq, S_LEN, 768, HID, stream);
  launch_gemm(hsb, WBkT, Bk, S_LEN, 256, HID, stream);
  launch_gemm(hsb, WBvT, Bv, S_LEN, 256, HID, stream);

  k_combine_q<<<(S_LEN * NH * 64   + T - 1) / T, T, 0, stream>>>(Aq, Bq, cs, sn, Qb);
  k_combine_k<<<(S_LEN * NKVH * 64 + T - 1) / T, T, 0, stream>>>(Ak, Bk, cs, sn, Kb);
  k_combine_v<<<(S_LEN * NKVH * HD + T - 1) / T, T, 0, stream>>>(Av, Bv, Vtb);

  k_attention<<<dim3(S_LEN / 128, NH), 256, 0, stream>>>(Qb, Kb, Vtb, Obuf);

  k_f32_to_bf16<<<(S_LEN * NH * HD + T - 1) / T, T, 0, stream>>>(Obuf, Obf, S_LEN * NH * HD);
  launch_gemm(Obf, WoT, (float*)d_out, S_LEN, HID, NH * HD, stream);
}
